// MambaBlock_816043786853
// MI455X (gfx1250) — matrix operand order repaired and checked by reading, not compiled, not run on hardware
//
#include <hip/hip_runtime.h>
#include <hip/hip_bf16.h>
#include <math.h>

// ---- model dims ----
#define DM   256      // D_MODEL
#define DI   512      // D_INNER
#define DS   16       // D_STATE
#define DTR  16       // DT_RANK
#define LL   1024     // H*W
#define BB   8        // batch
#define NT   (BB*LL)  // 8192 tokens

typedef __attribute__((ext_vector_type(16))) __bf16 v16bf;
typedef __attribute__((ext_vector_type(8)))  __bf16 v8bf;
typedef __attribute__((ext_vector_type(8)))  float  v8f;
typedef __attribute__((ext_vector_type(4)))  unsigned int v4u;
typedef __attribute__((ext_vector_type(8)))  int v8i;
typedef __attribute__((ext_vector_type(4)))  int v4i;

// ---- bf16 helpers (round-to-nearest-even) ----
__device__ __forceinline__ unsigned short f2bf(float f) {
  unsigned int u = __float_as_uint(f);
  unsigned int r = (u + 0x7FFFu + ((u >> 16) & 1u)) >> 16;
  return (unsigned short)r;
}
__device__ __forceinline__ float bf2f(unsigned short b) {
  return __uint_as_float(((unsigned int)b) << 16);
}
__device__ __forceinline__ float sigm(float x) { return 1.0f / (1.0f + expf(-x)); }

// =====================================================================
// TDM: DMA a 16-row x K-col bf16 panel (row-major, stride K) into LDS.
// Descriptor per CDNA5 ISA (2D tensor, groups 2/3 zero).
// =====================================================================
#if defined(__has_builtin)
#if __has_builtin(__builtin_amdgcn_tensor_load_to_lds)
#define HAVE_TDM 1
#endif
#endif

__device__ __forceinline__ void tdm_load_panel(const unsigned short* gptr,
                                               unsigned ldsByteOff, int K) {
#ifdef HAVE_TDM
  unsigned long long ga = (unsigned long long)(uintptr_t)gptr;
  v4u g0;
  g0.x = 1u;                                            // count=1 (valid D#)
  g0.y = ldsByteOff;                                    // lds_addr (bytes)
  g0.z = (unsigned)(ga & 0xFFFFFFFFu);                  // global_addr[31:0]
  g0.w = (unsigned)((ga >> 32) & 0x01FFFFFFu) | 0x80000000u; // addr[56:32] | type=2
  v8i g1;
  g1[0] = (int)(1u << 16);                              // data_size=1 -> 2 bytes
  g1[1] = (int)(((unsigned)K & 0xFFFFu) << 16);         // tensor_dim0[15:0]
  g1[2] = (int)(16u << 16);                             // tensor_dim0 hi=0 | tensor_dim1=16
  g1[3] = (int)(((unsigned)K & 0xFFFFu) << 16);         // tile_dim0 = K
  g1[4] = 16;                                           // tile_dim1=16, tile_dim2=0
  g1[5] = K;                                            // tensor_dim0_stride[31:0]
  g1[6] = 0;                                            // stride hi / dim1_stride lo
  g1[7] = 0;
  v4i z4 = {0, 0, 0, 0};
#if __clang_major__ >= 23
  v8i z8 = {0, 0, 0, 0, 0, 0, 0, 0};
  __builtin_amdgcn_tensor_load_to_lds(g0, g1, z4, z4, z8, 0);
#else
  __builtin_amdgcn_tensor_load_to_lds(g0, g1, z4, z4, 0);
#endif
  __builtin_amdgcn_s_wait_tensorcnt(0);
#else
  (void)gptr; (void)ldsByteOff; (void)K;
#endif
}

// =====================================================================
// Staged WMMA bf16 GEMM:  D[M,N] = A[M,K] @ W[N,K]^T
// Block = 8 waves, owns one 16-row M panel (staged in LDS via TDM);
// each wave holds NPW 16x16 accumulators (one A fragment feeds NPW WMMAs).
// Requires (N/16) % (8*NPW) == 0.
// mode 0: store bf16   mode 1: store f32   mode 2: f32 = softplus(d + bias[n])
// =====================================================================
template <int NPW>
__global__ void k_wmma_gemm_staged(const unsigned short* __restrict__ Abf,
                                   const unsigned short* __restrict__ Wbf,
                                   int M, int N, int K,
                                   int mode, const float* __restrict__ bias,
                                   float* __restrict__ outF,
                                   unsigned short* __restrict__ outB) {
  __shared__ __align__(16) unsigned short sA[16 * 512];   // 16KB, max K=512

  const int tilesN   = N >> 4;
  const int grpTiles = 8 * NPW;
  const int nGroups  = tilesN / grpTiles;
  const int tm  = (int)blockIdx.x / nGroups;
  const int grp = (int)blockIdx.x - tm * nGroups;
  const int wv  = (int)(threadIdx.x >> 5);
  const int lane = (int)(threadIdx.x & 31);
  const int kh = lane >> 4;
  const int lr = lane & 15;

  // ---- stage A panel (16 x K) into LDS ----
  const unsigned short* Apanel = Abf + (size_t)tm * 16 * K;
#ifdef HAVE_TDM
  if (threadIdx.x < 32) {
    tdm_load_panel(Apanel, (unsigned)(uintptr_t)sA, K);
  }
#else
  for (int i = threadIdx.x; i < (16 * K) / 8; i += 256) {
    int r = i / (K / 8), cb = i - r * (K / 8);
    *(v8bf*)(sA + r * K + cb * 8) = *(const v8bf*)(Apanel + (size_t)r * K + cb * 8);
  }
#endif
  __syncthreads();

  // ---- per-wave N tiles ----
  const int tn0 = grp * grpTiles + wv * NPW;
  const unsigned short* Wrow[NPW];
#pragma unroll
  for (int j = 0; j < NPW; ++j)
    Wrow[j] = Wbf + (size_t)((tn0 + j) * 16 + lr) * K;

  v8f acc[NPW];
#pragma unroll
  for (int j = 0; j < NPW; ++j) acc[j] = (v8f){0.f,0.f,0.f,0.f,0.f,0.f,0.f,0.f};

  for (int kt = 0; kt < K; kt += 32) {
    // A frag from LDS: elem e -> K = 16*kh + e
    union { v16bf v; v8bf h[2]; } a;
    a.h[0] = *(const v8bf*)(sA + lr * K + kt + kh * 16);
    a.h[1] = *(const v8bf*)(sA + lr * K + kt + kh * 16 + 8);
    if (kt + 32 < K) {
#pragma unroll
      for (int j = 0; j < NPW; ++j) __builtin_prefetch(Wrow[j] + kt + 32, 0, 0);
    }
#pragma unroll
    for (int j = 0; j < NPW; ++j) {
      // B frag: elem e -> K = 16*kh + e (contiguous 32B of weight row)
      v16bf bfrag = *(const v16bf*)(Wrow[j] + kt + kh * 16);
      acc[j] = __builtin_amdgcn_wmma_f32_16x16x32_bf16(
          false, a.v, false, bfrag, (short)0, acc[j], false, false);
    }
  }

#pragma unroll
  for (int j = 0; j < NPW; ++j) {
    int ncol = (tn0 + j) * 16 + lr;
#pragma unroll
    for (int v = 0; v < 8; ++v) {
      int m = tm * 16 + kh * 8 + v;          // C/D rows = 8*kh + v
      size_t oi = (size_t)m * N + ncol;
      float val = acc[j][v];
      if (mode == 0) {
        outB[oi] = f2bf(val);
      } else if (mode == 1) {
        outF[oi] = val;
      } else {
        val += bias[ncol];
        outF[oi] = (val > 20.f) ? val : log1pf(expf(val));   // softplus
      }
    }
  }
}

// =====================================================================
// Simple WMMA GEMM (one wave = one tile) for small-N x_proj (N=48)
// =====================================================================
__global__ void k_wmma_gemm(const unsigned short* __restrict__ Abf,
                            const unsigned short* __restrict__ Wbf,
                            int M, int N, int K, int tilesN,
                            float* __restrict__ outF) {
  int wave = (int)((blockIdx.x * blockDim.x + threadIdx.x) >> 5);
  int lane = (int)(threadIdx.x & 31);
  int tiles = (M >> 4) * tilesN;
  if (wave >= tiles) return;           // wave-uniform: EXEC stays all-1s
  int tm = wave / tilesN;
  int tn = wave - tm * tilesN;
  int kh = lane >> 4, lr = lane & 15;
  const unsigned short* Arow = Abf + (size_t)(tm * 16 + lr) * K;
  const unsigned short* Wrow = Wbf + (size_t)(tn * 16 + lr) * K;
  v8f acc = {0.f,0.f,0.f,0.f,0.f,0.f,0.f,0.f};
  for (int kt = 0; kt < K; kt += 32) {
    union { v16bf v; v8bf h[2]; } a;
    a.h[0] = *(const v8bf*)(Arow + kt + kh * 16);
    a.h[1] = *(const v8bf*)(Arow + kt + kh * 16 + 8);
    v16bf bfrag = *(const v16bf*)(Wrow + kt + kh * 16);
    if (kt + 32 < K) {
      __builtin_prefetch(Arow + kt + 32, 0, 0);
      __builtin_prefetch(Wrow + kt + 32, 0, 0);
    }
    acc = __builtin_amdgcn_wmma_f32_16x16x32_bf16(
        false, a.v, false, bfrag, (short)0, acc, false, false);
  }
  int ncol = tn * 16 + lr;
#pragma unroll
  for (int v = 0; v < 8; ++v) {
    int m = tm * 16 + kh * 8 + v;
    outF[(size_t)m * N + ncol] = acc[v];
  }
}

// ---- (B,C,H,W) -> (token, C) bf16 ----
__global__ void k_transpose_x(const float* __restrict__ x, unsigned short* __restrict__ xs) {
  int idx = blockIdx.x * blockDim.x + threadIdx.x;
  if (idx >= BB * DM * LL) return;
  int b = idx / (DM * LL);
  int c = (idx / LL) % DM;
  int l = idx % LL;
  xs[(size_t)(b * LL + l) * DM + c] = f2bf(x[idx]);
}

__global__ void k_f2bf(const float* __restrict__ src, unsigned short* __restrict__ dst, int n) {
  int i = blockIdx.x * blockDim.x + threadIdx.x;
  if (i < n) dst[i] = f2bf(src[i]);
}

// ---- dt_proj weight (512x16) -> zero-padded (512x32) bf16 ----
__global__ void k_pad_dtw(const float* __restrict__ src, unsigned short* __restrict__ dst) {
  int idx = blockIdx.x * blockDim.x + threadIdx.x;
  if (idx >= DI * 32) return;
  int d = idx >> 5, j = idx & 31;
  dst[idx] = (j < DTR) ? f2bf(src[d * DTR + j]) : (unsigned short)0;
}

// ---- depthwise causal conv (k=4, left pad 3) + bias + SiLU ----
__global__ void k_conv_silu(const unsigned short* __restrict__ xz,
                            const float* __restrict__ cw, const float* __restrict__ cb,
                            unsigned short* __restrict__ xc) {
  int idx = blockIdx.x * blockDim.x + threadIdx.x;
  if (idx >= NT * DI) return;
  int t = idx / DI, d = idx % DI;
  int b = t / LL, l = t % LL;
  float acc = cb[d];
#pragma unroll
  for (int j = 0; j < 4; ++j) {
    int ll = l - 3 + j;
    if (ll >= 0) acc += cw[d * 4 + j] * bf2f(xz[(size_t)(b * LL + ll) * (2 * DI) + d]);
  }
  acc = acc * sigm(acc);               // SiLU
  xc[idx] = f2bf(acc);
}

// ---- dt slice of x_dbl -> zero-padded (NT x 32) bf16 ----
__global__ void k_dtpad(const float* __restrict__ xdbl, unsigned short* __restrict__ dtp) {
  int idx = blockIdx.x * blockDim.x + threadIdx.x;
  if (idx >= NT * 32) return;
  int t = idx >> 5, j = idx & 31;
  dtp[idx] = (j < DTR) ? f2bf(xdbl[(size_t)t * 48 + j]) : (unsigned short)0;
}

// ---- selective scan: one thread per (b,d), 16 states in registers;
//      fused +u*D and output gating y *= silu(z) ----
__global__ void k_scan(const unsigned short* __restrict__ xc,
                       const float* __restrict__ dtf,
                       const float* __restrict__ xdbl,
                       const float* __restrict__ A_log,
                       const float* __restrict__ ssm_D,
                       const unsigned short* __restrict__ xz,
                       unsigned short* __restrict__ ybf) {
  int tid = blockIdx.x * blockDim.x + threadIdx.x;
  if (tid >= BB * DI) return;
  int b = tid / DI, d = tid % DI;
  float A[DS], h[DS];
#pragma unroll
  for (int n = 0; n < DS; ++n) { A[n] = -expf(A_log[d * DS + n]); h[n] = 0.f; }
  float Dp = ssm_D[d];
  for (int l = 0; l < LL; ++l) {
    int t = b * LL + l;
    float dt = dtf[(size_t)t * DI + d];
    float u = bf2f(xc[(size_t)t * DI + d]);
    const float* bc = xdbl + (size_t)t * 48;
    float y = 0.f;
#pragma unroll
    for (int n = 0; n < DS; ++n) {
      h[n] = expf(dt * A[n]) * h[n] + dt * bc[DTR + n] * u;
      y += h[n] * bc[DTR + DS + n];
    }
    y += u * Dp;
    float z = bf2f(xz[(size_t)t * (2 * DI) + DI + d]);
    y *= z * sigm(z);                  // y * silu(z)
    ybf[(size_t)t * DI + d] = f2bf(y);
  }
}

// ---- LayerNorm over channels (256), one block per token ----
__global__ void k_layernorm(float* __restrict__ out,
                            const float* __restrict__ lw, const float* __restrict__ lb) {
  __shared__ float red[DM];
  int t = blockIdx.x, c = threadIdx.x;
  float v = out[(size_t)t * DM + c];
  red[c] = v; __syncthreads();
  for (int s = DM / 2; s > 0; s >>= 1) { if (c < s) red[c] += red[c + s]; __syncthreads(); }
  float mu = red[0] / DM; __syncthreads();
  float dv = v - mu;
  red[c] = dv * dv; __syncthreads();
  for (int s = DM / 2; s > 0; s >>= 1) { if (c < s) red[c] += red[c + s]; __syncthreads(); }
  float var = red[0] / DM; __syncthreads();
  out[(size_t)t * DM + c] = dv * rsqrtf(var + 1e-5f) * lw[c] + lb[c];
}

// ---- spatial mean per (b,c) ----
__global__ void k_spatial_mean(const float* __restrict__ out, float* __restrict__ s0) {
  __shared__ float red[256];
  int bc = blockIdx.x;
  int b = bc >> 8, c = bc & 255;
  int tid = threadIdx.x;
  float acc = 0.f;
  for (int l = tid; l < LL; l += 256) acc += out[(size_t)(b * LL + l) * DM + c];
  red[tid] = acc; __syncthreads();
  for (int s = 128; s > 0; s >>= 1) { if (tid < s) red[tid] += red[tid + s]; __syncthreads(); }
  if (tid == 0) s0[bc] = red[0] / (float)LL;
}

// ---- SE MLP: sigmoid(relu(s0 @ w1^T) @ w2^T), one block per batch ----
__global__ void k_se(const float* __restrict__ s0, const float* __restrict__ w1,
                     const float* __restrict__ w2, float* __restrict__ s) {
  __shared__ float sh[DM];
  __shared__ float hid[DM / 16];
  int b = blockIdx.x, t = threadIdx.x;
  sh[t] = s0[b * DM + t];
  __syncthreads();
  if (t < DM / 16) {
    float a = 0.f;
    for (int c = 0; c < DM; ++c) a += w1[t * DM + c] * sh[c];
    hid[t] = fmaxf(a, 0.f);
  }
  __syncthreads();
  float a = 0.f;
#pragma unroll
  for (int j = 0; j < DM / 16; ++j) a += w2[t * (DM / 16) + j] * hid[j];
  s[b * DM + t] = sigm(a);
}

// ---- out = ln_out * s + identity, back to (B,C,H,W) ----
__global__ void k_final(const float* __restrict__ lnout, const float* __restrict__ s,
                        const float* __restrict__ x, float* __restrict__ dout) {
  int idx = blockIdx.x * blockDim.x + threadIdx.x;
  if (idx >= BB * DM * LL) return;
  int b = idx / (DM * LL);
  int c = (idx / LL) % DM;
  int l = idx % LL;
  dout[idx] = lnout[(size_t)(b * LL + l) * DM + c] * s[b * DM + c] + x[idx];
}

// ---- workspace layout (bytes) ----
static const size_t OFF_XS   = 0;
static const size_t OFF_WIN  = OFF_XS   + (size_t)NT * DM * 2;
static const size_t OFF_WXP  = OFF_WIN  + (size_t)(2 * DI) * DM * 2;
static const size_t OFF_WDT  = OFF_WXP  + (size_t)48 * DI * 2;
static const size_t OFF_WOUT = OFF_WDT  + (size_t)DI * 32 * 2;
static const size_t OFF_XZ   = OFF_WOUT + (size_t)DM * DI * 2;
static const size_t OFF_XC   = OFF_XZ   + (size_t)NT * 2 * DI * 2;
static const size_t OFF_XDBL = OFF_XC   + (size_t)NT * DI * 2;
static const size_t OFF_DTP  = OFF_XDBL + (size_t)NT * 48 * 4;
static const size_t OFF_DT   = OFF_DTP  + (size_t)NT * 32 * 2;
static const size_t OFF_YBF  = OFF_DT   + (size_t)NT * DI * 4;
static const size_t OFF_OUT  = OFF_YBF  + (size_t)NT * DI * 2;
static const size_t OFF_S0   = OFF_OUT  + (size_t)NT * DM * 4;
static const size_t OFF_S    = OFF_S0   + (size_t)BB * DM * 4;

extern "C" void kernel_launch(void* const* d_in, const int* in_sizes, int n_in,
                              void* d_out, int out_size, void* d_ws, size_t ws_size,
                              hipStream_t stream) {
  const float* x      = (const float*)d_in[0];
  const float* in_w   = (const float*)d_in[1];
  const float* conv_w = (const float*)d_in[2];
  const float* conv_b = (const float*)d_in[3];
  const float* xp_w   = (const float*)d_in[4];
  const float* dt_w   = (const float*)d_in[5];
  const float* dt_b   = (const float*)d_in[6];
  const float* A_log  = (const float*)d_in[7];
  const float* ssm_D  = (const float*)d_in[8];
  const float* out_w  = (const float*)d_in[9];
  const float* ln_w   = (const float*)d_in[10];
  const float* ln_b   = (const float*)d_in[11];
  const float* se_w1  = (const float*)d_in[12];
  const float* se_w2  = (const float*)d_in[13];
  float* out = (float*)d_out;

  char* ws = (char*)d_ws;
  unsigned short* xs_bf   = (unsigned short*)(ws + OFF_XS);
  unsigned short* win_bf  = (unsigned short*)(ws + OFF_WIN);
  unsigned short* wxp_bf  = (unsigned short*)(ws + OFF_WXP);
  unsigned short* wdt_bf  = (unsigned short*)(ws + OFF_WDT);
  unsigned short* wout_bf = (unsigned short*)(ws + OFF_WOUT);
  unsigned short* xz_bf   = (unsigned short*)(ws + OFF_XZ);
  unsigned short* xc_bf   = (unsigned short*)(ws + OFF_XC);
  float*          xdbl    = (float*)(ws + OFF_XDBL);
  unsigned short* dtp_bf  = (unsigned short*)(ws + OFF_DTP);
  float*          dt_f    = (float*)(ws + OFF_DT);
  unsigned short* y_bf    = (unsigned short*)(ws + OFF_YBF);
  float*          out_f   = (float*)(ws + OFF_OUT);
  float*          s0_f    = (float*)(ws + OFF_S0);
  float*          s_f     = (float*)(ws + OFF_S);

  const int TB = 256;
  auto blk = [](long long n, int t) { return (unsigned)((n + t - 1) / t); };
  auto staged_blocks = [](int M, int N, int npw) {
    return (unsigned)((M / 16) * ((N / 16) / (8 * npw)));
  };

  // 1) convert/transpose inputs to bf16
  k_transpose_x<<<blk((long long)BB * DM * LL, TB), TB, 0, stream>>>(x, xs_bf);
  k_f2bf<<<blk(2 * DI * DM, TB), TB, 0, stream>>>(in_w, win_bf, 2 * DI * DM);
  k_f2bf<<<blk(48 * DI, TB), TB, 0, stream>>>(xp_w, wxp_bf, 48 * DI);
  k_pad_dtw<<<blk(DI * 32, TB), TB, 0, stream>>>(dt_w, wdt_bf);
  k_f2bf<<<blk(DM * DI, TB), TB, 0, stream>>>(out_w, wout_bf, DM * DI);

  // 2) in_proj GEMM (TDM-staged, 4 N-tiles/wave): (8192x256)@(256x1024) -> xz bf16
  k_wmma_gemm_staged<4><<<staged_blocks(NT, 2 * DI, 4), TB, 0, stream>>>(
      xs_bf, win_bf, NT, 2 * DI, DM, 0, nullptr, nullptr, xz_bf);

  // 3) depthwise causal conv + SiLU
  k_conv_silu<<<blk((long long)NT * DI, TB), TB, 0, stream>>>(xz_bf, conv_w, conv_b, xc_bf);

  // 4) x_proj GEMM (small N=48, simple path): (8192x512)@(512x48) -> x_dbl f32
  k_wmma_gemm<<<blk((long long)(NT / 16) * 3 * 32, TB), TB, 0, stream>>>(
      xc_bf, wxp_bf, NT, 48, DI, 3, xdbl);

  // 5) dt pad + dt_proj GEMM (TDM-staged, K padded to 32) + bias + softplus
  k_dtpad<<<blk((long long)NT * 32, TB), TB, 0, stream>>>(xdbl, dtp_bf);
  k_wmma_gemm_staged<4><<<staged_blocks(NT, DI, 4), TB, 0, stream>>>(
      dtp_bf, wdt_bf, NT, DI, 32, 2, dt_b, dt_f, nullptr);

  // 6) selective scan (+u*D, *silu(z)) -> y bf16
  k_scan<<<blk(BB * DI, TB), TB, 0, stream>>>(xc_bf, dt_f, xdbl, A_log, ssm_D, xz_bf, y_bf);

  // 7) out_proj GEMM (TDM-staged, 2 N-tiles/wave): (8192x512)@(512x256) -> out f32
  k_wmma_gemm_staged<2><<<staged_blocks(NT, DM, 2), TB, 0, stream>>>(
      y_bf, wout_bf, NT, DM, DI, 1, nullptr, out_f, nullptr);

  // 8) LayerNorm, SE, residual
  k_layernorm<<<NT, DM, 0, stream>>>(out_f, ln_w, ln_b);
  k_spatial_mean<<<BB * DM, 256, 0, stream>>>(out_f, s0_f);
  k_se<<<BB, DM, 0, stream>>>(s0_f, se_w1, se_w2, s_f);
  k_final<<<blk((long long)BB * DM * LL, TB), TB, 0, stream>>>(out_f, s_f, x, out);
}
